// TransformerBlock_3393024164190
// MI455X (gfx1250) — matrix operand order repaired and checked by reading, not compiled, not run on hardware
//
#include <hip/hip_runtime.h>
#include <hip/hip_bf16.h>

typedef __attribute__((ext_vector_type(16))) __bf16 v16bf;
typedef __attribute__((ext_vector_type(8)))  __bf16 v8bf;
typedef __attribute__((ext_vector_type(2)))  __bf16 v2bf;
typedef __attribute__((ext_vector_type(8)))  float  v8f;
typedef unsigned int u32x4 __attribute__((ext_vector_type(4)));
typedef int          i32x4 __attribute__((ext_vector_type(4)));
typedef int          i32x8 __attribute__((ext_vector_type(8)));

#define BM 64
#define BN 64
#define BK 32

#if __has_builtin(__builtin_amdgcn_tensor_load_to_lds) && \
    __has_builtin(__builtin_amdgcn_s_wait_tensorcnt)
#define HAVE_TDM 1
#endif

__device__ __forceinline__ v8f wmma_bf16(v16bf a, v16bf b, v8f c) {
  return __builtin_amdgcn_wmma_f32_16x16x32_bf16(false, a, false, b, (short)0, c,
                                                 false, false);
}

__device__ __forceinline__ v16bf frag_from_2(const __bf16* p0, const __bf16* p1) {
  v8bf lo = *(const v8bf*)p0;   // ds_load_b128
  v8bf hi = *(const v8bf*)p1;   // ds_load_b128
  return __builtin_shufflevector(lo, hi, 0, 1, 2, 3, 4, 5, 6, 7, 8, 9, 10, 11,
                                 12, 13, 14, 15);
}

// A-fragment from row-major LDS tile [16][ldk] (K contiguous).
__device__ __forceinline__ v16bf frag_a_vec(const __bf16* As, int ldk, int lane) {
  int m = lane & 15, kb = (lane >> 4) << 3;
  const __bf16* base = As + m * ldk + kb;
  return frag_from_2(base, base + 16);
}

// B-fragment from K-contiguous (N-major) LDS tile Bst[n*ldk + k].
__device__ __forceinline__ v16bf frag_b_vec(const __bf16* Bst, int ldk, int n0,
                                            int lane) {
  int n = n0 + (lane & 15), kb = (lane >> 4) << 3;
  const __bf16* base = Bst + n * ldk + kb;
  return frag_from_2(base, base + 16);
}

#ifdef HAVE_TDM
// Issue one TDM 2D tile load: gsrc is bf16, row-major [rows][strideElems],
// tile = tileRows x rowElems, rows >= rowsValid zero-filled (OOB).
// D# packing per CDNA5 ISA 8.3/8.4 (group0 128b, group1 256b).
__device__ __forceinline__ void tdm_load_2d_bf16(const __bf16* gsrc,
                                                 __bf16* ldst, int rowElems,
                                                 int rowsValid, int tileRows,
                                                 long strideElems) {
  unsigned lds = (unsigned)(unsigned long)(uintptr_t)ldst;  // LDS byte offset
  unsigned long ga = (unsigned long)(uintptr_t)gsrc;
  u32x4 g0;
  g0[0] = 1u;                                   // count=1, user descriptor
  g0[1] = lds;                                  // lds_addr
  g0[2] = (unsigned)(ga & 0xffffffffu);         // global_addr[31:0]
  g0[3] = (unsigned)((ga >> 32) & 0x01ffffffu)  // global_addr[56:32]
          | (2u << 30);                         // type=2 (image)
  unsigned long td0 = (unsigned long)rowElems;  // tensor_dim0
  unsigned long td1 = (unsigned long)rowsValid; // tensor_dim1
  unsigned long st0 = (unsigned long)strideElems;
  i32x8 g1;
  g1[0] = (int)(1u << 16);                                   // data_size=2B
  g1[1] = (int)((td0 & 0xffffu) << 16);                      // dim0[15:0]
  g1[2] = (int)(((td0 >> 16) & 0xffffu) | ((td1 & 0xffffu) << 16));
  g1[3] = (int)(((td1 >> 16) & 0xffffu) |
                (((unsigned)rowElems & 0xffffu) << 16));     // tile_dim0
  g1[4] = (int)((unsigned)tileRows & 0xffffu);               // tile_dim1
  g1[5] = (int)(st0 & 0xffffffffu);                          // dim0_stride lo
  g1[6] = (int)((st0 >> 32) & 0xffffu);                      // dim0_stride hi
  g1[7] = 0;
  i32x4 gz = {0, 0, 0, 0};
#if __clang_major__ >= 23
  i32x8 gz8 = {0, 0, 0, 0, 0, 0, 0, 0};
  __builtin_amdgcn_tensor_load_to_lds(g0, g1, gz, gz, gz8, 0);
#else
  __builtin_amdgcn_tensor_load_to_lds(g0, g1, gz, gz, 0);
#endif
}
#endif

enum { EPI_NONE = 0, EPI_SILU = 1, EPI_ADD = 2 };

// ---------------------------------------------------------------------------
// Dense GEMM: C[M,N] = epilogue(A[M,K] @ B[K,N]); A fp32 row-major,
// B pre-converted bf16 N-major (Bt[n*K+k]).  K % 32 == 0.
// ---------------------------------------------------------------------------
template <int EPI>
__global__ __launch_bounds__(256) void gemm_wmma(
    const float* __restrict__ A, const __bf16* __restrict__ Bt,
    float* __restrict__ C, const float* __restrict__ R,
    int M, int N, int K, int lda, int ldc) {
  __shared__ alignas(16) __bf16 As[BM * BK];   // row-major [m][k]
  __shared__ alignas(16) __bf16 Bst[BN * BK];  // [n][k] K-contiguous
  const int tid = threadIdx.x, lane = tid & 31, wv = tid >> 5;
  const int wm = wv >> 1, wn = wv & 1;
  const int m0 = blockIdx.x * BM, n0 = blockIdx.y * BN;
  v8f acc0 = {}, acc1 = {};
  for (int k0 = 0; k0 < K; k0 += BK) {
#ifdef HAVE_TDM
    if (wv == 0)  // one DMA per block; TDM ignores EXEC, wave-level issue
      tdm_load_2d_bf16(Bt + (long)n0 * K + k0, Bst, BK,
                       (N - n0 < BN) ? (N - n0) : BN, BN, (long)K);
#else
    {  // manual 2D block copy fallback (16B chunks, 1 per thread)
      int n = tid >> 2, ko = (tid & 3) << 3;
      v8bf ch;
#pragma unroll
      for (int z = 0; z < 8; ++z) ch[z] = (__bf16)0.f;
      if (n0 + n < N) ch = *(const v8bf*)(Bt + (long)(n0 + n) * K + k0 + ko);
      *(v8bf*)(Bst + n * BK + ko) = ch;
    }
#endif
    // stage A (64x32): fp32 -> bf16 pairs, coalesced float2 reads
#pragma unroll
    for (int i = 0; i < 4; ++i) {
      int idx = tid + i * 256;
      int r = idx >> 4, k = (idx & 15) * 2;
      float2 v = make_float2(0.f, 0.f);
      if (m0 + r < M) v = *(const float2*)(A + (long)(m0 + r) * lda + k0 + k);
      v2bf pk;
      pk[0] = (__bf16)v.x;
      pk[1] = (__bf16)v.y;
      *(v2bf*)(As + r * BK + k) = pk;
    }
    if (k0 + BK < K)  // global_prefetch_b8 of next A tile
      __builtin_prefetch(A + (long)(m0 + (tid >> 2)) * lda + k0 + BK, 0, 1);
#ifdef HAVE_TDM
    if (wv == 0) __builtin_amdgcn_s_wait_tensorcnt(0);
#endif
    __syncthreads();
    v16bf a = frag_a_vec(As + wm * 16 * BK, BK, lane);
    v16bf b0 = frag_b_vec(Bst, BK, wn * 32, lane);
    v16bf b1 = frag_b_vec(Bst, BK, wn * 32 + 16, lane);
    acc0 = wmma_bf16(a, b0, acc0);
    acc1 = wmma_bf16(a, b1, acc1);
    __syncthreads();
  }
  const int half = lane >> 4, col = lane & 15;
#pragma unroll
  for (int r = 0; r < 8; ++r) {
    int gm = m0 + wm * 16 + half * 8 + r;
    if (gm >= M) continue;
#pragma unroll
    for (int t = 0; t < 2; ++t) {
      int gn = n0 + wn * 32 + t * 16 + col;
      if (gn >= N) continue;
      float v = t ? acc1[r] : acc0[r];
      if (EPI == EPI_SILU) v = v / (1.f + __expf(-v));
      if (EPI == EPI_ADD) v += R[(long)gm * ldc + gn];
      C[(long)gm * ldc + gn] = v;
    }
  }
}

// ---------------------------------------------------------------------------
// MoE expert GEMM with token gather/scatter; weights bf16 N-major per expert.
// ---------------------------------------------------------------------------
template <bool UP>
__global__ __launch_bounds__(256) void moe_gemm(
    const float* __restrict__ X, const __bf16* __restrict__ Wt,
    float* __restrict__ C, const int* __restrict__ tokl,
    const float* __restrict__ gatel, const int* __restrict__ counts,
    const int* __restrict__ offs, int N, int K, int ldx, int ldc) {
  __shared__ alignas(16) __bf16 As[BM * BK];
  __shared__ alignas(16) __bf16 Bst[BN * BK];
  const int e = blockIdx.z;
  const int cnt = counts[e];
  const int m0 = blockIdx.x * BM;
  if (m0 >= cnt) return;
  const int base = offs[e];
  const __bf16* We = Wt + (long)e * K * N;
  const int tid = threadIdx.x, lane = tid & 31, wv = tid >> 5;
  const int wm = wv >> 1, wn = wv & 1;
  const int n0 = blockIdx.y * BN;
  v8f acc0 = {}, acc1 = {};
  for (int k0 = 0; k0 < K; k0 += BK) {
#ifdef HAVE_TDM
    if (wv == 0)
      tdm_load_2d_bf16(We + (long)n0 * K + k0, Bst, BK, BN, BN, (long)K);
#else
    {
      int n = tid >> 2, ko = (tid & 3) << 3;
      v8bf ch = *(const v8bf*)(We + (long)(n0 + n) * K + k0 + ko);
      *(v8bf*)(Bst + n * BK + ko) = ch;
    }
#endif
#pragma unroll
    for (int i = 0; i < 4; ++i) {
      int idx = tid + i * 256;
      int r = idx >> 4, k = (idx & 15) * 2;
      float2 v = make_float2(0.f, 0.f);
      if (m0 + r < cnt) {
        long g = base + m0 + r;
        long arow = UP ? (long)(tokl[g] >> 1) : g;
        v = *(const float2*)(X + arow * ldx + k0 + k);
      }
      v2bf pk;
      pk[0] = (__bf16)v.x;
      pk[1] = (__bf16)v.y;
      *(v2bf*)(As + r * BK + k) = pk;
    }
#ifdef HAVE_TDM
    if (wv == 0) __builtin_amdgcn_s_wait_tensorcnt(0);
#endif
    __syncthreads();
    v16bf a = frag_a_vec(As + wm * 16 * BK, BK, lane);
    v16bf b0 = frag_b_vec(Bst, BK, wn * 32, lane);
    v16bf b1 = frag_b_vec(Bst, BK, wn * 32 + 16, lane);
    acc0 = wmma_bf16(a, b0, acc0);
    acc1 = wmma_bf16(a, b1, acc1);
    __syncthreads();
  }
  const int half = lane >> 4, col = lane & 15;
#pragma unroll
  for (int r = 0; r < 8; ++r) {
    int gi = m0 + wm * 16 + half * 8 + r;
    if (gi >= cnt) continue;
    long g = base + gi;
#pragma unroll
    for (int t = 0; t < 2; ++t) {
      int gn = n0 + wn * 32 + t * 16 + col;
      if (gn >= N) continue;
      float v = t ? acc1[r] : acc0[r];
      if (UP) {
        v = v / (1.f + __expf(-v));
        C[g * (long)ldc + gn] = v;
      } else {
        int ts = tokl[g];  // token*2 + slot (unique per gathered row)
        C[(long)ts * ldc + gn] = gatel[g] * v;
      }
    }
  }
}

// ---------------------------------------------------------------------------
// Flash attention per (b,h). 8 waves x 16 query rows, 32-key tiles.
// ---------------------------------------------------------------------------
__global__ __launch_bounds__(256) void mla_attention(
    const float* __restrict__ qb, const float* __restrict__ kvup,
    const float* __restrict__ kr, float* __restrict__ attno, int S) {
  __shared__ alignas(16) __bf16 Ktw[32 * 96];      // [key][d]  d-contiguous
  __shared__ alignas(16) __bf16 Vst[128 * 32];     // [d][key]  key-contiguous
  __shared__ alignas(16) __bf16 Ps[8][16 * 32];    // per-wave P, row-major
  const int tid = threadIdx.x, lane = tid & 31, wv = tid >> 5;
  const int half = lane >> 4, col = lane & 15;
  const int qblk = blockIdx.x, bh = blockIdx.y;
  const int b = bh >> 2, h = bh & 3;
  const long tb = (long)b * S;
  const int qrow0 = qblk * 128 + wv * 16;
  const float scale = 0.10206207261596575f;  // 1/sqrt(96)

  v16bf qa[3];
  {
    int m = lane & 15, kb = half * 8;
    const float* qrow = qb + (tb + qrow0 + m) * 384 + h * 96;
#pragma unroll
    for (int ks = 0; ks < 3; ++ks)
#pragma unroll
      for (int g = 0; g < 2; ++g) {
        int d0 = ks * 32 + g * 16 + kb;
        float4 fa = *(const float4*)(qrow + d0);
        float4 fb = *(const float4*)(qrow + d0 + 4);
        qa[ks][g * 8 + 0] = (__bf16)(fa.x * scale);
        qa[ks][g * 8 + 1] = (__bf16)(fa.y * scale);
        qa[ks][g * 8 + 2] = (__bf16)(fa.z * scale);
        qa[ks][g * 8 + 3] = (__bf16)(fa.w * scale);
        qa[ks][g * 8 + 4] = (__bf16)(fb.x * scale);
        qa[ks][g * 8 + 5] = (__bf16)(fb.y * scale);
        qa[ks][g * 8 + 6] = (__bf16)(fb.z * scale);
        qa[ks][g * 8 + 7] = (__bf16)(fb.w * scale);
      }
  }

  v8f of[8];
#pragma unroll
  for (int i = 0; i < 8; ++i) of[i] = (v8f){};
  float mst[8], lst[8];
#pragma unroll
  for (int r = 0; r < 8; ++r) { mst[r] = -3.0e38f; lst[r] = 0.f; }

  const int kend = (qblk + 1) * 128;  // causal block bound
  for (int kb0 = 0; kb0 < kend; kb0 += 32) {
    for (int i = tid; i < 32 * 48; i += 256) {
      int key = i / 48, d = (i % 48) * 2;
      long t = tb + kb0 + key;
      float v0, v1;
      if (d < 64) {
        v0 = kvup[t * 768 + h * 192 + d];
        v1 = kvup[t * 768 + h * 192 + d + 1];
      } else {
        v0 = kr[t * 32 + (d - 64)];
        v1 = kr[t * 32 + (d - 63)];
      }
      v2bf pk;
      pk[0] = (__bf16)v0;
      pk[1] = (__bf16)v1;
      *(v2bf*)(Ktw + key * 96 + d) = pk;
    }
    for (int i = tid; i < 128 * 16; i += 256) {
      int d = i & 127, key = (i >> 7) * 2;
      long t = tb + kb0 + key;
      float v0 = kvup[t * 768 + h * 192 + 64 + d];
      float v1 = kvup[(t + 1) * 768 + h * 192 + 64 + d];
      v2bf pk;
      pk[0] = (__bf16)v0;
      pk[1] = (__bf16)v1;
      *(v2bf*)(Vst + d * 32 + key) = pk;
    }
    __syncthreads();

    v8f s0 = {}, s1 = {};
#pragma unroll
    for (int ks = 0; ks < 3; ++ks) {
      v16bf b0 = frag_b_vec(Ktw + ks * 32, 96, 0, lane);
      v16bf b1 = frag_b_vec(Ktw + ks * 32, 96, 16, lane);
      s0 = wmma_bf16(qa[ks], b0, s0);
      s1 = wmma_bf16(qa[ks], b1, s1);
    }
#pragma unroll
    for (int r = 0; r < 8; ++r) {
      int qi = qrow0 + half * 8 + r;
      float v0 = (kb0 + col <= qi) ? s0[r] : -3.0e38f;
      float v1 = (kb0 + 16 + col <= qi) ? s1[r] : -3.0e38f;
      float mt = fmaxf(v0, v1);
#pragma unroll
      for (int xm = 8; xm >= 1; xm >>= 1) mt = fmaxf(mt, __shfl_xor(mt, xm, 32));
      float mn = fmaxf(mst[r], mt);
      float alpha = __expf(mst[r] - mn);
      float p0 = (v0 < -1.0e37f) ? 0.f : __expf(v0 - mn);
      float p1 = (v1 < -1.0e37f) ? 0.f : __expf(v1 - mn);
      float ps = p0 + p1;
#pragma unroll
      for (int xm = 8; xm >= 1; xm >>= 1) ps += __shfl_xor(ps, xm, 32);
      lst[r] = lst[r] * alpha + ps;
      mst[r] = mn;
#pragma unroll
      for (int f = 0; f < 8; ++f) of[f][r] *= alpha;
      int M = half * 8 + r;
      Ps[wv][M * 32 + col] = (__bf16)p0;
      Ps[wv][M * 32 + 16 + col] = (__bf16)p1;
    }
    __syncthreads();
    v16bf pa = frag_a_vec(Ps[wv], 32, lane);
#pragma unroll
    for (int dn = 0; dn < 8; ++dn) {
      v16bf vb = frag_b_vec(Vst, 32, dn * 16, lane);
      of[dn] = wmma_bf16(pa, vb, of[dn]);
    }
    __syncthreads();
  }
#pragma unroll
  for (int r = 0; r < 8; ++r) {
    long t = tb + qrow0 + half * 8 + r;
    float inv = 1.f / lst[r];
#pragma unroll
    for (int dn = 0; dn < 8; ++dn)
      attno[t * 512 + h * 128 + dn * 16 + col] = of[dn][r] * inv;
  }
}

// ---------------------------------------------------------------------------
// Elementwise / small kernels
// ---------------------------------------------------------------------------
__global__ void convtrans_kernel(const float* __restrict__ W,
                                 __bf16* __restrict__ Wt, int K, int N) {
  // W: [K][N] fp32 (per batch) -> Wt: [N][K] bf16
  const float* Wb = W + (long)blockIdx.y * K * N;
  __bf16* Wtb = Wt + (long)blockIdx.y * K * N;
  int i = blockIdx.x * blockDim.x + threadIdx.x;
  if (i >= K * N) return;
  int n = i / K, k = i - n * K;
  Wtb[(long)n * K + k] = (__bf16)Wb[(long)k * N + n];
}

__global__ void rmsnorm_kernel(const float* __restrict__ in,
                               const float* __restrict__ w,
                               float* __restrict__ out, int T, int cols,
                               int instride, int outstride) {
  int row = blockIdx.x * 8 + (threadIdx.x >> 5);
  int lane = threadIdx.x & 31;
  if (row >= T) return;
  const float* xr = in + (long)row * instride;
  float ss = 0.f;
  for (int c = lane; c < cols; c += 32) { float v = xr[c]; ss += v * v; }
#pragma unroll
  for (int m = 16; m >= 1; m >>= 1) ss += __shfl_xor(ss, m, 32);
  float rr = rsqrtf(ss / (float)cols + 1e-6f);
  float* orow = out + (long)row * outstride;
  for (int c = lane; c < cols; c += 32) orow[c] = xr[c] * rr * w[c];
}

__global__ void rope_q_kernel(float* __restrict__ q, int T, int S) {
  int i = blockIdx.x * blockDim.x + threadIdx.x;
  if (i >= T * 4 * 16) return;
  int f = i & 15, th = i >> 4;
  int h = th & 3, t = th >> 2;
  int s = t & (S - 1);
  float invf = __powf(10000.f, -(float)(2 * f) / 32.f);
  float ang = (float)s * invf, c = __cosf(ang), sn = __sinf(ang);
  float* p = q + (long)t * 384 + h * 96 + 64;
  float x1 = p[f], x2 = p[f + 16];
  p[f] = x1 * c - x2 * sn;
  p[f + 16] = x2 * c + x1 * sn;
}

__global__ void rope_k_kernel(const float* __restrict__ ckv,
                              float* __restrict__ kr, int T, int S) {
  int i = blockIdx.x * blockDim.x + threadIdx.x;
  if (i >= T * 16) return;
  int f = i & 15, t = i >> 4;
  int s = t & (S - 1);
  float invf = __powf(10000.f, -(float)(2 * f) / 32.f);
  float ang = (float)s * invf, c = __cosf(ang), sn = __sinf(ang);
  float x1 = ckv[(long)t * 160 + 128 + f];
  float x2 = ckv[(long)t * 160 + 144 + f];
  kr[(long)t * 32 + f] = x1 * c - x2 * sn;
  kr[(long)t * 32 + 16 + f] = x2 * c + x1 * sn;
}

__global__ void gate_kernel(const float* __restrict__ x2,
                            const float* __restrict__ Wg,
                            int* __restrict__ topi, float* __restrict__ topw,
                            int T) {
  int t = blockIdx.x * blockDim.x + threadIdx.x;
  if (t >= T) return;
  float acc[8];
#pragma unroll
  for (int e = 0; e < 8; ++e) acc[e] = 0.f;
  const float* xr = x2 + (long)t * 512;
  for (int d = 0; d < 512; ++d) {
    float xv = xr[d];
#pragma unroll
    for (int e = 0; e < 8; ++e) acc[e] += xv * Wg[d * 8 + e];
  }
  float mx = acc[0];
#pragma unroll
  for (int e = 1; e < 8; ++e) mx = fmaxf(mx, acc[e]);
  float s = 0.f;
#pragma unroll
  for (int e = 0; e < 8; ++e) { acc[e] = __expf(acc[e] - mx); s += acc[e]; }
#pragma unroll
  for (int e = 0; e < 8; ++e) acc[e] /= s;
  int i0 = 0;
#pragma unroll
  for (int e = 1; e < 8; ++e) if (acc[e] > acc[i0]) i0 = e;
  int i1 = (i0 == 0) ? 1 : 0;
#pragma unroll
  for (int e = 0; e < 8; ++e)
    if (e != i0 && acc[e] > acc[i1]) i1 = e;
  float w0 = acc[i0], w1 = acc[i1], sw = w0 + w1;
  topi[t * 2] = i0;
  topi[t * 2 + 1] = i1;
  topw[t * 2] = w0 / sw;
  topw[t * 2 + 1] = w1 / sw;
}

__global__ void route_count(const int* __restrict__ topi, int* counts, int n) {
  int i = blockIdx.x * blockDim.x + threadIdx.x;
  if (i < n) atomicAdd(&counts[topi[i]], 1);
}

__global__ void route_scan(const int* __restrict__ counts, int* offs,
                           int* cursor) {
  if (threadIdx.x == 0 && blockIdx.x == 0) {
    int s = 0;
    for (int e = 0; e < 8; ++e) { offs[e] = s; cursor[e] = s; s += counts[e]; }
  }
}

__global__ void route_scatter(const int* __restrict__ topi,
                              const float* __restrict__ topw, int* cursor,
                              int* __restrict__ tokl,
                              float* __restrict__ gatel, int n) {
  int i = blockIdx.x * blockDim.x + threadIdx.x;
  if (i >= n) return;
  int e = topi[i];
  int pos = atomicAdd(&cursor[e], 1);
  tokl[pos] = i;  // token*2 + slot
  gatel[pos] = topw[i];
}

__global__ void combine_kernel(const float* __restrict__ x2,
                               const float* __restrict__ y2,
                               const float* __restrict__ sh,
                               float* __restrict__ outx, int n) {
  int i = blockIdx.x * blockDim.x + threadIdx.x;
  if (i >= n) return;
  long t = i >> 9;
  int c = i & 511;
  outx[i] = x2[i] + sh[i] + y2[(t * 2) * 512 + c] + y2[(t * 2 + 1) * 512 + c];
}

// ---------------------------------------------------------------------------
extern "C" void kernel_launch(void* const* d_in, const int* in_sizes, int n_in,
                              void* d_out, int out_size, void* d_ws,
                              size_t ws_size, hipStream_t stream) {
  (void)in_sizes; (void)n_in; (void)out_size; (void)ws_size;
  const float* x    = (const float*)d_in[0];
  const float* n2w  = (const float*)d_in[1];
  const float* Wdq  = (const float*)d_in[2];
  const float* qnw  = (const float*)d_in[3];
  const float* Wuq  = (const float*)d_in[4];
  const float* Wdkv = (const float*)d_in[5];
  const float* kvnw = (const float*)d_in[6];
  const float* Wukv = (const float*)d_in[7];
  const float* Wo   = (const float*)d_in[8];
  const float* Wg   = (const float*)d_in[9];
  const float* We1  = (const float*)d_in[10];
  const float* We2  = (const float*)d_in[11];
  const float* Ws1  = (const float*)d_in[12];
  const float* Ws2  = (const float*)d_in[13];

  const long T = 8192;
  const int S = 2048;
  float* outx = (float*)d_out;
  float* kvout = outx + T * 512;  // ckv_full output (pre-norm, pre-rope)

  // Workspace layout (region A reused by MoE buffers after attention).
  float* F = (float*)d_ws;
  float* cq = F;                   // T*192
  float* ckvn = cq + T * 192;      // T*128
  float* kr = ckvn + T * 128;      // T*32
  float* qb = kr + T * 32;         // T*384
  float* kvup = qb + T * 384;      // T*768
  float* attno = kvup + T * 768;   // T*512
  float* h1 = attno + T * 512;     // T*512
  float* regEnd = h1 + T * 512;    // = F + T*2528
  float* hbuf = F;                 // reuse region A: 2T*1024 <= T*2528
  float* hs = F;                   // reuse after hbuf dead: T*1024
  float* x2 = regEnd;              // T*512
  float* y2 = x2 + T * 512;        // 2T*512
  float* shacc = y2 + 2 * T * 512; // T*512
  float* topw = shacc + T * 512;   // 2T
  float* gatel = topw + 2 * T;     // 2T
  int* topi = (int*)(gatel + 2 * T);
  int* tokl = topi + 2 * T;
  int* counts = tokl + 2 * T;
  int* offs = counts + 8;
  int* cursor = offs + 8;
  // bf16 transposed weight cache (persists through the call)
  uintptr_t wbase = ((uintptr_t)(cursor + 8) + 15) & ~(uintptr_t)15;
  __bf16* Wdq_t  = (__bf16*)wbase;              // 512*192
  __bf16* Wdkv_t = Wdq_t + 512 * 192;           // 512*160
  __bf16* Wuq_t  = Wdkv_t + 512 * 160;          // 192*384
  __bf16* Wukv_t = Wuq_t + 192 * 384;           // 128*768
  __bf16* Wo_t   = Wukv_t + 128 * 768;          // 512*512
  __bf16* We1_t  = Wo_t + 512 * 512;            // 8*512*1024
  __bf16* We2_t  = We1_t + 8L * 512 * 1024;     // 8*1024*512
  __bf16* Ws1_t  = We2_t + 8L * 1024 * 512;     // 2*512*1024
  __bf16* Ws2_t  = Ws1_t + 2L * 512 * 1024;     // 2*1024*512

  dim3 blk(256);

  // One-time (per call) weight convert+transpose to bf16 N-major.
  convtrans_kernel<<<dim3((512 * 192 + 255) / 256, 1), blk, 0, stream>>>(Wdq, Wdq_t, 512, 192);
  convtrans_kernel<<<dim3((512 * 160 + 255) / 256, 1), blk, 0, stream>>>(Wdkv, Wdkv_t, 512, 160);
  convtrans_kernel<<<dim3((192 * 384 + 255) / 256, 1), blk, 0, stream>>>(Wuq, Wuq_t, 192, 384);
  convtrans_kernel<<<dim3((128 * 768 + 255) / 256, 1), blk, 0, stream>>>(Wukv, Wukv_t, 128, 768);
  convtrans_kernel<<<dim3((512 * 512 + 255) / 256, 1), blk, 0, stream>>>(Wo, Wo_t, 512, 512);
  convtrans_kernel<<<dim3((512 * 1024 + 255) / 256, 8), blk, 0, stream>>>(We1, We1_t, 512, 1024);
  convtrans_kernel<<<dim3((1024 * 512 + 255) / 256, 8), blk, 0, stream>>>(We2, We2_t, 1024, 512);
  convtrans_kernel<<<dim3((512 * 1024 + 255) / 256, 2), blk, 0, stream>>>(Ws1, Ws1_t, 512, 1024);
  convtrans_kernel<<<dim3((1024 * 512 + 255) / 256, 2), blk, 0, stream>>>(Ws2, Ws2_t, 1024, 512);

  // cq = rmsnorm(x @ W_dq)
  gemm_wmma<EPI_NONE><<<dim3(128, 3), blk, 0, stream>>>(x, Wdq_t, cq, nullptr,
                                                        (int)T, 192, 512, 512, 192);
  rmsnorm_kernel<<<dim3((int)(T / 8)), blk, 0, stream>>>(cq, qnw, cq, (int)T, 192, 192, 192);
  // ckv_full = x @ W_dkv  (straight into d_out kv slice)
  gemm_wmma<EPI_NONE><<<dim3(128, 3), blk, 0, stream>>>(x, Wdkv_t, kvout, nullptr,
                                                        (int)T, 160, 512, 512, 160);
  // c_kv norm and k_rope rope
  rmsnorm_kernel<<<dim3((int)(T / 8)), blk, 0, stream>>>(kvout, kvnw, ckvn, (int)T, 128, 160, 128);
  rope_k_kernel<<<dim3((int)(T * 16 / 256)), blk, 0, stream>>>(kvout, kr, (int)T, S);
  // q = cq @ W_uq, rope in place
  gemm_wmma<EPI_NONE><<<dim3(128, 6), blk, 0, stream>>>(cq, Wuq_t, qb, nullptr,
                                                        (int)T, 384, 192, 192, 384);
  rope_q_kernel<<<dim3((int)(T * 64 / 256)), blk, 0, stream>>>(qb, (int)T, S);
  // kv_up = rmsnorm(c_kv) @ W_ukv
  gemm_wmma<EPI_NONE><<<dim3(128, 12), blk, 0, stream>>>(ckvn, Wukv_t, kvup, nullptr,
                                                         (int)T, 768, 128, 128, 768);
  // attention
  mla_attention<<<dim3(16, 16), blk, 0, stream>>>(qb, kvup, kr, attno, S);
  // h1 = attn_out @ W_o + x ; x2 = rmsnorm(h1)
  gemm_wmma<EPI_ADD><<<dim3(128, 8), blk, 0, stream>>>(attno, Wo_t, h1, x,
                                                       (int)T, 512, 512, 512, 512);
  rmsnorm_kernel<<<dim3((int)(T / 8)), blk, 0, stream>>>(h1, n2w, x2, (int)T, 512, 512, 512);
  // MoE gating + routing
  gate_kernel<<<dim3((int)(T / 256)), blk, 0, stream>>>(x2, Wg, topi, topw, (int)T);
  hipMemsetAsync(counts, 0, 8 * sizeof(int), stream);
  route_count<<<dim3((int)(2 * T / 256)), blk, 0, stream>>>(topi, counts, (int)(2 * T));
  route_scan<<<1, 32, 0, stream>>>(counts, offs, cursor);
  route_scatter<<<dim3((int)(2 * T / 256)), blk, 0, stream>>>(topi, topw, cursor,
                                                              tokl, gatel, (int)(2 * T));
  // routed experts: gathered up (silu) then down (scatter per (token,slot))
  moe_gemm<true><<<dim3(128, 16, 8), blk, 0, stream>>>(x2, We1_t, hbuf, tokl, gatel,
                                                       counts, offs, 1024, 512, 512, 1024);
  moe_gemm<false><<<dim3(128, 8, 8), blk, 0, stream>>>(hbuf, We2_t, y2, tokl, gatel,
                                                       counts, offs, 512, 1024, 1024, 512);
  // shared experts (sequential over k, accumulate in shacc)
  for (int k = 0; k < 2; ++k) {
    gemm_wmma<EPI_SILU><<<dim3(128, 16), blk, 0, stream>>>(
        x2, Ws1_t + (long)k * 512 * 1024, hs, nullptr, (int)T, 1024, 512, 512, 1024);
    if (k == 0)
      gemm_wmma<EPI_NONE><<<dim3(128, 8), blk, 0, stream>>>(
          hs, Ws2_t + (long)k * 1024 * 512, shacc, nullptr, (int)T, 512, 1024, 1024, 512);
    else
      gemm_wmma<EPI_ADD><<<dim3(128, 8), blk, 0, stream>>>(
          hs, Ws2_t + (long)k * 1024 * 512, shacc, shacc, (int)T, 512, 1024, 1024, 512);
  }
  // final: out = x2 + routed + shared
  combine_kernel<<<dim3((int)(T * 512 / 256)), blk, 0, stream>>>(
      x2, y2, shacc, outx, (int)(T * 512));
}
